// NodeUpdateModule_74526272520922
// MI455X (gfx1250) — matrix operand order repaired and checked by reading, not compiled, not run on hardware
//
#include <hip/hip_runtime.h>

typedef __attribute__((ext_vector_type(16))) _Float16 v16h;
typedef __attribute__((ext_vector_type(8)))  _Float16 v8h;
typedef __attribute__((ext_vector_type(8)))  float    v8f;

#define DEV __device__ __forceinline__

constexpr int D_IN   = 544;           // 16+128+128+128+144 (loss column folded into bias)
constexpr int KB0    = 17;            // 544 = 17 k-blocks of 32
constexpr int KB1    = 8;             // K=256 for layers 1/2
constexpr int NT256  = 16;            // 256 cols = 16 n-tiles
constexpr int NTOUT  = 9;             // 144 cols = 9 n-tiles

// packed f16 weight fragment sizes (halves). Layout per (kb,nt):
// 32 lanes x 16 halves, lane-major: element e of lane L is
// W[K = kb*32 + 16*(L/16) + e][N = nt*16 + (L%16)]  (contiguous k per lane half, same k order as load_a)
constexpr int WB0_H = KB0 * NT256 * 512;   // 139264
constexpr int WB1_H = KB1 * NT256 * 512;   //  65536
constexpr int WB2_H = KB1 * NTOUT * 512;   //  36864
constexpr int WB_TOT = WB0_H + WB1_H + WB2_H;   // 241664 halves = 483328 B

// ---------------------------------------------------------------------------
// Weight repacking: f32 row-major -> f16 WMMA-B fragment layout
// ---------------------------------------------------------------------------
__global__ __launch_bounds__(256) void prep_weights(const float* __restrict__ W0,
                                                    const float* __restrict__ W1,
                                                    const float* __restrict__ W2,
                                                    _Float16* __restrict__ wb) {
  int idx = blockIdx.x * 256 + threadIdx.x;
  if (idx >= WB_TOT) return;
  const float* W; int KSRC, NSRC, NT; _Float16* dst; int local;
  if (idx < WB0_H)               { W = W0; KSRC = 544; NSRC = 256; NT = NT256; dst = wb;                 local = idx; }
  else if (idx < WB0_H + WB1_H)  { W = W1; KSRC = 256; NSRC = 256; NT = NT256; dst = wb + WB0_H;         local = idx - WB0_H; }
  else                           { W = W2; KSRC = 256; NSRC = 144; NT = NTOUT; dst = wb + WB0_H + WB1_H; local = idx - WB0_H - WB1_H; }
  int e    = local & 15;
  int lane = (local >> 4) & 31;
  int nt   = (local >> 9) % NT;
  int kb   = local / (512 * NT);
  int K  = kb * 32 + ((lane >> 4) << 4) + e;
  int Nc = nt * 16 + (lane & 15);
  float v = (K < KSRC) ? W[K * NSRC + Nc] : 0.0f;
  dst[local] = (_Float16)v;
}

// effective layer-0 bias: b0 + ln_loss_b[0] * W0[row 544]  (exact fold of the
// G=1 layernorm column: (x-mu) == 0 -> output == bias, constant for all rows)
__global__ __launch_bounds__(256) void prep_bias(const float* __restrict__ W0,
                                                 const float* __restrict__ b0,
                                                 const float* __restrict__ ln_loss_b,
                                                 float* __restrict__ b0e) {
  int n = threadIdx.x;
  b0e[n] = b0[n] + ln_loss_b[0] * W0[544 * 256 + n];
}

// ---------------------------------------------------------------------------
// helpers
// ---------------------------------------------------------------------------

// layernorm of one G-element group for TWO rows at once: half-wave 0 (lanes
// 0-15) does row+0, half-wave 1 (lanes 16-31) does row+1.
DEV void ln_group2(const float* __restrict__ src, int srcStride, int G,
                   const float* __restrict__ sc, const float* __restrict__ bi,
                   _Float16* dst, int dstStride, int sub, int half) {
  const float* s_ = src + half * srcStride;
  _Float16*    d_ = dst + half * dstStride;
  float v[9];
  int cnt = 0;
  float s = 0.f, q = 0.f;
  for (int i = sub; i < G; i += 16) {
    float x = s_[i];
    v[cnt++] = x;
    s += x; q += x * x;
  }
#pragma unroll
  for (int m = 1; m < 16; m <<= 1) { s += __shfl_xor(s, m, 16); q += __shfl_xor(q, m, 16); }
  float invG = 1.0f / (float)G;
  float mu   = s * invG;
  float var  = q * invG - mu * mu;
  float rinv = rsqrtf(var + 1e-5f);
  int c = 0;
  for (int i = sub; i < G; i += 16) {
    d_[i] = (_Float16)((v[c] - mu) * rinv * sc[i] + bi[i]);
    ++c;
  }
}

DEV v16h cat8(v8h lo, v8h hi) {
  return __builtin_shufflevector(lo, hi, 0, 1, 2, 3, 4, 5, 6, 7,
                                 8, 9, 10, 11, 12, 13, 14, 15);
}

// A fragment (16x32 f16): lane holds row M=lane%16, half=lane/16;
// halves 0..15 = K[kb*32 + 16*half + 0..15]. Two 16B LDS loads.
DEV v16h load_a(const _Float16* rowbase, int stride, int kb, int lanelo, int half) {
  const _Float16* ap = rowbase + lanelo * stride + kb * 32 + half * 16;
  return cat8(*(const v8h*)ap, *(const v8h*)(ap + 8));
}

// B fragment from packed weights: 32B contiguous per lane
DEV v16h load_b(const _Float16* frag, int lane) {
  const _Float16* bp = frag + lane * 16;
  return cat8(*(const v8h*)bp, *(const v8h*)(bp + 8));
}

// prefetch an entire B k-slab (nLines x 128B) one k-iteration ahead:
// 32 lanes each pull distinct cachelines -> global_prefetch_b8
DEV void prefetch_slab(const _Float16* base, int lane, int nIter) {
  const char* p = (const char*)base + lane * 128;
  for (int i = 0; i < nIter; ++i)
    __builtin_prefetch(p + i * 32 * 128, 0, 3);
}

// LN (256 cols) + ReLU on one accumulator set, store 16x256 f16 tile to LDS
DEV void ln_relu_store(const v8f* acc, const float* pb, const float* ps,
                       const float* pbb, _Float16* hdst, int lanelo, int half) {
  float mu[8], rinv[8];
#pragma unroll
  for (int r = 0; r < 8; ++r) {
    float s = 0.f, q = 0.f;
#pragma unroll
    for (int nt = 0; nt < 16; ++nt) { float v = acc[nt][r] + pb[nt]; s += v; q += v * v; }
#pragma unroll
    for (int m = 1; m < 16; m <<= 1) { s += __shfl_xor(s, m, 16); q += __shfl_xor(q, m, 16); }
    float m_  = s * (1.f / 256.f);
    float var = q * (1.f / 256.f) - m_ * m_;
    mu[r] = m_; rinv[r] = rsqrtf(var + 1e-5f);
  }
#pragma unroll
  for (int nt = 0; nt < 16; ++nt) {
    int c = nt * 16 + lanelo;
#pragma unroll
    for (int r = 0; r < 8; ++r) {
      float v = acc[nt][r] + pb[nt];
      v = (v - mu[r]) * rinv[r] * ps[nt] + pbb[nt];
      v = fmaxf(v, 0.f);
      hdst[(r + 8 * half) * 256 + c] = (_Float16)v;
    }
  }
}

// ---------------------------------------------------------------------------
// fused kernel: one wave == TWO 16-row tiles (M=32) -> each B fragment fetched
// once per wave feeds two WMMAs (halves redundant L2 weight traffic)
// ---------------------------------------------------------------------------
__global__ __launch_bounds__(32) void fused_node_update(
    const float* __restrict__ logits, const float* __restrict__ hidden,
    const float* __restrict__ layer_pe, const float* __restrict__ intra_pe,
    const int* __restrict__ layer, const float* __restrict__ msg,
    const float* __restrict__ ln_logits_s, const float* __restrict__ ln_logits_b,
    const float* __restrict__ ln_hidden_s, const float* __restrict__ ln_hidden_b,
    const float* __restrict__ ln_lpe_s,    const float* __restrict__ ln_lpe_b,
    const float* __restrict__ ln_ipe_s,    const float* __restrict__ ln_ipe_b,
    const float* __restrict__ ln_msg_s,    const float* __restrict__ ln_msg_b,
    const float* __restrict__ b0e, const float* __restrict__ s0, const float* __restrict__ bb0,
    const float* __restrict__ b1,  const float* __restrict__ s1, const float* __restrict__ bb1,
    const float* __restrict__ b2,
    const _Float16* __restrict__ wb0, const _Float16* __restrict__ wb1,
    const _Float16* __restrict__ wb2,
    float* __restrict__ out, int numTiles) {
  __shared__ _Float16 xl[2 * 16 * D_IN];   // 34816 B; reused as two 16x256 h-tiles

  const int lane   = threadIdx.x;          // 0..31
  const int half   = lane >> 4;
  const int lanelo = lane & 15;
  const int sub    = lanelo;

  const int tile0 = blockIdx.x * 2;
  const int tile1 = (tile0 + 1 < numTiles) ? tile0 + 1 : numTiles - 1;
  const int rowT[2] = { tile0 * 16, tile1 * 16 };

  const v8f vzero = {0.f, 0.f, 0.f, 0.f, 0.f, 0.f, 0.f, 0.f};

  // warm up the first layer-0 B slab while we do layernorms
  prefetch_slab(wb0, lane, 4);

  // ---- Stage 1: 5 layernorms per row -> f16 x tiles (16 x 544 each) ----
#pragma unroll
  for (int t = 0; t < 2; ++t) {
    _Float16* xt = xl + t * (16 * D_IN);
    for (int rp = 0; rp < 8; ++rp) {
      const int row = rowT[t] + rp * 2;       // half-wave adds +1
      _Float16* xr = xt + rp * 2 * D_IN;
      ln_group2(logits   + row * 16,  16,  16,  ln_logits_s, ln_logits_b, xr + 0,   D_IN, sub, half);
      ln_group2(hidden   + row * 128, 128, 128, ln_hidden_s, ln_hidden_b, xr + 16,  D_IN, sub, half);
      ln_group2(layer_pe + row * 128, 128, 128, ln_lpe_s,    ln_lpe_b,    xr + 144, D_IN, sub, half);
      ln_group2(intra_pe + row * 128, 128, 128, ln_ipe_s,    ln_ipe_b,    xr + 272, D_IN, sub, half);
      ln_group2(msg      + row * 144, 144, 144, ln_msg_s,    ln_msg_b,    xr + 400, D_IN, sub, half);
    }
  }
  __syncthreads();

  // ---- Layer 0: x(32x544) @ W0(544x256), B shared across both M-tiles ----
  v8f acc0[16], acc1[16];
#pragma unroll
  for (int nt = 0; nt < 16; ++nt) { acc0[nt] = vzero; acc1[nt] = vzero; }
  for (int kb = 0; kb < KB0; ++kb) {
    const _Float16* wk = wb0 + kb * (NT256 * 512);
    if (kb + 1 < KB0) prefetch_slab(wk + NT256 * 512, lane, 4);   // next 16KB slab
    v16h a0 = load_a(xl,               D_IN, kb, lanelo, half);
    v16h a1 = load_a(xl + 16 * D_IN,   D_IN, kb, lanelo, half);
#pragma unroll
    for (int nt = 0; nt < 16; ++nt) {
      v16h b = load_b(wk + nt * 512, lane);
      acc0[nt] = __builtin_amdgcn_wmma_f32_16x16x32_f16(false, a0, false, b,
                                                        (short)0, acc0[nt], false, false);
      acc1[nt] = __builtin_amdgcn_wmma_f32_16x16x32_f16(false, a1, false, b,
                                                        (short)0, acc1[nt], false, false);
    }
  }
  __syncthreads();

  // ---- LN0 + ReLU -> h0 tiles (16x256 f16) ----
  {
    float pb[16], ps[16], pbb[16];
#pragma unroll
    for (int nt = 0; nt < 16; ++nt) {
      int c = nt * 16 + lanelo;
      pb[nt] = b0e[c]; ps[nt] = s0[c]; pbb[nt] = bb0[c];
    }
    ln_relu_store(acc0, pb, ps, pbb, xl,               lanelo, half);
    ln_relu_store(acc1, pb, ps, pbb, xl + 16 * D_IN,   lanelo, half);
  }
  prefetch_slab(wb1, lane, 4);
  __syncthreads();

  // ---- Layer 1: h0(32x256) @ W1(256x256) ----
#pragma unroll
  for (int nt = 0; nt < 16; ++nt) { acc0[nt] = vzero; acc1[nt] = vzero; }
  for (int kb = 0; kb < KB1; ++kb) {
    const _Float16* wk = wb1 + kb * (NT256 * 512);
    if (kb + 1 < KB1) prefetch_slab(wk + NT256 * 512, lane, 4);
    v16h a0 = load_a(xl,             256, kb, lanelo, half);
    v16h a1 = load_a(xl + 16 * D_IN, 256, kb, lanelo, half);
#pragma unroll
    for (int nt = 0; nt < 16; ++nt) {
      v16h b = load_b(wk + nt * 512, lane);
      acc0[nt] = __builtin_amdgcn_wmma_f32_16x16x32_f16(false, a0, false, b,
                                                        (short)0, acc0[nt], false, false);
      acc1[nt] = __builtin_amdgcn_wmma_f32_16x16x32_f16(false, a1, false, b,
                                                        (short)0, acc1[nt], false, false);
    }
  }
  __syncthreads();

  // ---- LN1 + ReLU -> h1 tiles ----
  {
    float pb[16], ps[16], pbb[16];
#pragma unroll
    for (int nt = 0; nt < 16; ++nt) {
      int c = nt * 16 + lanelo;
      pb[nt] = b1[c]; ps[nt] = s1[c]; pbb[nt] = bb1[c];
    }
    ln_relu_store(acc0, pb, ps, pbb, xl,               lanelo, half);
    ln_relu_store(acc1, pb, ps, pbb, xl + 16 * D_IN,   lanelo, half);
  }
  prefetch_slab(wb2, lane, 3);
  __syncthreads();

  // ---- Layer 2: h1(32x256) @ W2(256x144) ----
  v8f c20[9], c21[9];
#pragma unroll
  for (int nt = 0; nt < 9; ++nt) { c20[nt] = vzero; c21[nt] = vzero; }
  for (int kb = 0; kb < KB1; ++kb) {
    const _Float16* wk = wb2 + kb * (NTOUT * 512);
    if (kb + 1 < KB1) prefetch_slab(wk + NTOUT * 512, lane, 3);   // next 9KB slab
    v16h a0 = load_a(xl,             256, kb, lanelo, half);
    v16h a1 = load_a(xl + 16 * D_IN, 256, kb, lanelo, half);
#pragma unroll
    for (int nt = 0; nt < 9; ++nt) {
      v16h b = load_b(wk + nt * 512, lane);
      c20[nt] = __builtin_amdgcn_wmma_f32_16x16x32_f16(false, a0, false, b,
                                                       (short)0, c20[nt], false, false);
      c21[nt] = __builtin_amdgcn_wmma_f32_16x16x32_f16(false, a1, false, b,
                                                       (short)0, c21[nt], false, false);
    }
  }

  // ---- Output: delta + b2, masked residual into [logits | hidden] ----
  {
    float pb2[9];
#pragma unroll
    for (int nt = 0; nt < 9; ++nt) pb2[nt] = b2[nt * 16 + lanelo];
#pragma unroll
    for (int t = 0; t < 2; ++t) {
      const v8f* cc = t ? c21 : c20;
      int lv[8];
#pragma unroll
      for (int r = 0; r < 8; ++r) lv[r] = layer[rowT[t] + r + 8 * half];
#pragma unroll
      for (int nt = 0; nt < 9; ++nt) {
        int c = nt * 16 + lanelo;
#pragma unroll
        for (int r = 0; r < 8; ++r) {
          int row    = rowT[t] + r + 8 * half;
          float d    = cc[nt][r] + pb2[nt];
          float base = (nt == 0) ? logits[row * 16 + c]
                                 : hidden[row * 128 + (c - 16)];
          float o = (lv[r] > 0) ? base + d : base;
          __builtin_nontemporal_store(o, &out[row * 144 + c]);
        }
      }
    }
  }
}

// ---------------------------------------------------------------------------
extern "C" void kernel_launch(void* const* d_in, const int* in_sizes, int n_in,
                              void* d_out, int out_size, void* d_ws, size_t ws_size,
                              hipStream_t stream) {
  const float* logits   = (const float*)d_in[0];
  const float* hidden   = (const float*)d_in[1];
  const float* layer_pe = (const float*)d_in[2];
  const float* intra_pe = (const float*)d_in[3];
  const int*   layer    = (const int*)d_in[5];
  const float* msg      = (const float*)d_in[6];
  const float* ln_logits_s = (const float*)d_in[7];
  const float* ln_logits_b = (const float*)d_in[8];
  const float* ln_hidden_s = (const float*)d_in[9];
  const float* ln_hidden_b = (const float*)d_in[10];
  const float* ln_lpe_s    = (const float*)d_in[11];
  const float* ln_lpe_b    = (const float*)d_in[12];
  const float* ln_ipe_s    = (const float*)d_in[13];
  const float* ln_ipe_b    = (const float*)d_in[14];
  const float* ln_loss_b   = (const float*)d_in[16];
  const float* ln_msg_s    = (const float*)d_in[17];
  const float* ln_msg_b    = (const float*)d_in[18];
  const float* W0  = (const float*)d_in[19];
  const float* b0  = (const float*)d_in[20];
  const float* s0  = (const float*)d_in[21];
  const float* bb0 = (const float*)d_in[22];
  const float* W1  = (const float*)d_in[23];
  const float* b1  = (const float*)d_in[24];
  const float* s1  = (const float*)d_in[25];
  const float* bb1 = (const float*)d_in[26];
  const float* W2  = (const float*)d_in[27];
  const float* b2  = (const float*)d_in[28];

  _Float16* wb  = (_Float16*)d_ws;
  float*    b0e = (float*)(wb + WB_TOT);     // 256 f32 right after packed weights

  const int N = in_sizes[0] / 16;            // rows (200000)
  const int numTiles = (N + 15) / 16;        // 12500
  const int numBlocks = (numTiles + 1) / 2;  // 2 tiles per wave/block

  prep_weights<<<(WB_TOT + 255) / 256, 256, 0, stream>>>(W0, W1, W2, wb);
  prep_bias<<<1, 256, 0, stream>>>(W0, b0, ln_loss_b, b0e);
  fused_node_update<<<numBlocks, 32, 0, stream>>>(
      logits, hidden, layer_pe, intra_pe, layer, msg,
      ln_logits_s, ln_logits_b, ln_hidden_s, ln_hidden_b,
      ln_lpe_s, ln_lpe_b, ln_ipe_s, ln_ipe_b,
      ln_msg_s, ln_msg_b,
      b0e, s0, bb0, b1, s1, bb1, b2,
      wb, wb + WB0_H, wb + WB0_H + WB1_H,
      (float*)d_out, numTiles);
}
